// AdvancedKoopmanModel_17609365913720
// MI455X (gfx1250) — matrix operand order repaired and checked by reading, not compiled, not run on hardware
//
#include <hip/hip_runtime.h>

typedef __attribute__((ext_vector_type(16))) _Float16 v16h;
typedef __attribute__((ext_vector_type(8)))  float    v8f;
typedef __attribute__((ext_vector_type(4)))  int      v4i;

#define NT 8192     // nodes / rollout steps
#define NE 131072   // edges
#define KDIM 256    // koopman dim

#ifndef __has_builtin
#define __has_builtin(x) 0
#endif
#if __has_builtin(__builtin_amdgcn_global_load_async_to_lds_b128)
#define USE_ASYNC_GATHER 1
#define AS1 __attribute__((address_space(1)))
#define AS3 __attribute__((address_space(3)))
#else
#define USE_ASYNC_GATHER 0
#endif

// ---------------------------------------------------------------------------
// WMMA helpers (wave32, V_WMMA_F32_16X16X32_F16)
// A: 16x32 f16, row = lane&15; lane<16 holds K 0..7,16..23; lane>=16: 8..15,24..31
// ap must already point at (row, grp*8) within the A tile.
// ---------------------------------------------------------------------------
__device__ __forceinline__ v16h frag_a_at(const _Float16* r) {
  v16h a;
  ((v4i*)&a)[0] = *(const v4i*)(r);        // K = grp*8 + 0..7
  ((v4i*)&a)[1] = *(const v4i*)(r + 16);   // K = 16 + grp*8 + 0..7
  return a;
}

__device__ __forceinline__ v8f wmma_f16(v16h a, v16h b, v8f c) {
  return __builtin_amdgcn_wmma_f32_16x16x32_f16(false, a, false, b, (short)0, c,
                                                false, false);
}

__device__ __forceinline__ void atomic_add_f32(float* p, float v) {
  __hip_atomic_fetch_add(p, v, __ATOMIC_RELAXED, __HIP_MEMORY_SCOPE_AGENT);
}

// ---------------------------------------------------------------------------
// Pack weight W[N, Kin] (row major f32) into per-lane B-fragment layout:
// P[((kt*ntn + nt)*32 + lane)*16 + i] = B[kt*32 + (lane>>4)*8 + (i&7) + (i>>3)*16][nt*16 + (lane&15)]
// where B[k][n] = W[n][k]; zero pad for k >= Kin. Each lane's 16 halves are
// contiguous (32B) -> B fragment load is a single v16h read.
// ---------------------------------------------------------------------------
__global__ void k_pack_w(const float* __restrict__ W, _Float16* __restrict__ P,
                         int N, int Kin, int Kpad) {
  int idx = blockIdx.x * 256 + threadIdx.x;
  int total = Kpad * N;
  if (idx >= total) return;
  int i    = idx & 15;
  int lane = (idx >> 4) & 31;
  int tile = idx >> 9;
  int ntn  = N >> 4;
  int kt = tile / ntn, nt = tile - kt * ntn;
  int k = kt * 32 + ((lane >> 4) << 3) + (i & 7) + ((i >> 3) << 4);
  int n = nt * 16 + (lane & 15);
  P[idx] = (k < Kin) ? (_Float16)W[n * Kin + k] : (_Float16)0.0f;
}

__global__ void k_cvt16(const float* __restrict__ x, _Float16* __restrict__ y, int n) {
  int i = blockIdx.x * 256 + threadIdx.x;
  if (i < n) y[i] = (_Float16)x[i];
}

// ---------------------------------------------------------------------------
// Edge message MLP + scatter-add (segment_sum over dst).
// Per block: 16 edges. Stage1: gather cat(x[dst],x[src],ea) -> LDS f16 tile
//   (async global->LDS b128 when available, else b128 via VGPRs).
// Stage2: hidden = relu(cat @ WaT + ba)  (WMMA, f16 back to LDS)
// Stage3: msg = hidden @ WbT + bb -> atomicAdd into acc[dst].
// 128 threads = 4 waves; each wave owns pairs of 16-wide N tiles (A reuse).
// Note: Kpad = 2*Cin + 32 always (Cin multiple of 32), so tail is exactly 32.
// ---------------------------------------------------------------------------
__global__ void __launch_bounds__(128) k_edge_conv(
    const _Float16* __restrict__ xn, int Cin,
    const int* __restrict__ src, const int* __restrict__ dst,
    const float* __restrict__ ea,
    const _Float16* __restrict__ WaP, const float* __restrict__ ba, int K1pad, int Cmid,
    const _Float16* __restrict__ WbP, const float* __restrict__ bb, int Cout,
    float* __restrict__ acc) {
  __shared__ __align__(16) _Float16 s_cat[16 * 544];
  __shared__ __align__(16) _Float16 s_h[16 * 256];
  __shared__ int s_dst[16];
  __shared__ int s_src[16];

  const int tid = threadIdx.x;
  const int e0  = blockIdx.x << 4;
  if (tid < 16) {
    s_dst[tid] = dst[e0 + tid];
    s_src[tid] = src[e0 + tid];
  }
  __syncthreads();

  const int C2 = Cin * 2;
  // ---- stage 1: gather (16B chunks; rows are contiguous) ----
  const int rowChunks = Cin >> 3;                 // 16B chunks per source row
  const int totalChunks = 16 * 2 * rowChunks;
  for (int c = tid; c < totalChunks; c += 128) {
    const int le    = c / (2 * rowChunks);
    const int rc    = c - le * (2 * rowChunks);
    const int isSrc = rc >= rowChunks;
    const int ck    = isSrc ? rc - rowChunks : rc;
    const int node  = isSrc ? s_src[le] : s_dst[le];
    const _Float16* gp = xn + (size_t)node * Cin + (ck << 3);
    _Float16* lp = s_cat + le * K1pad + (isSrc ? Cin : 0) + (ck << 3);
#if USE_ASYNC_GATHER
    __builtin_amdgcn_global_load_async_to_lds_b128(
        (AS1 v4i*)(AS1 void*)gp, (AS3 v4i*)(AS3 void*)lp, 0, 0);
#else
    *(v4i*)lp = *(const v4i*)gp;
#endif
  }
  // tail: 4 edge-attr halves + 28 zero-pad halves per edge (32 total)
  for (int idx = tid; idx < 16 * 32; idx += 128) {
    const int le = idx >> 5, k2 = idx & 31;
    _Float16 val = (_Float16)0.0f;
    if (k2 < 4) val = (_Float16)ea[(e0 + le) * 4 + k2];
    s_cat[le * K1pad + C2 + k2] = val;
  }
#if USE_ASYNC_GATHER
  asm volatile("s_wait_asynccnt 0x0" ::: "memory");
#endif
  __syncthreads();

  const int wave = tid >> 5, lane = tid & 31;
  const int col = lane & 15, grp = lane >> 4;
  const int aoff = (lane & 15);           // A row
  const int asub = ((lane >> 4) << 3);    // grp*8

  // ---- stage 2: hidden = relu(cat @ WaT + ba) ----
  {
    const int ntm = Cmid >> 4, ntk = K1pad >> 5, half = ntm >> 1;
    const int bstep = ntm * 32, boff = half * 32;
    for (int nt0 = wave; nt0 < half; nt0 += 4) {
      const int nt1 = nt0 + half;
      v8f c0, c1;
      const float b0v = ba[nt0 * 16 + col];
      const float b1v = ba[nt1 * 16 + col];
#pragma unroll
      for (int v = 0; v < 8; ++v) { c0[v] = b0v; c1[v] = b1v; }
      const _Float16* ap = s_cat + aoff * K1pad + asub;
      const v16h* bp = (const v16h*)WaP + (size_t)nt0 * 32 + lane;
      for (int kt = 0; kt < ntk; ++kt) {
        v16h a = frag_a_at(ap);
        c0 = wmma_f16(a, bp[0], c0);
        c1 = wmma_f16(a, bp[boff], c1);
        ap += 32;
        bp += bstep;
      }
#pragma unroll
      for (int v = 0; v < 8; ++v) {
        const int rr = (v + (grp << 3)) * Cmid + col;
        s_h[rr + nt0 * 16] = (_Float16)fmaxf(c0[v], 0.0f);
        s_h[rr + nt1 * 16] = (_Float16)fmaxf(c1[v], 0.0f);
      }
    }
  }
  __syncthreads();

  // ---- stage 3: msg = hidden @ WbT + bb ; scatter-add over dst ----
  {
    const int ntn = Cout >> 4, ntk = Cmid >> 5, half = ntn >> 1;
    const int bstep = ntn * 32, boff = half * 32;
    for (int nt0 = wave; nt0 < half; nt0 += 4) {
      const int nt1 = nt0 + half;
      v8f c0, c1;
      const float b0v = bb[nt0 * 16 + col];
      const float b1v = bb[nt1 * 16 + col];
#pragma unroll
      for (int v = 0; v < 8; ++v) { c0[v] = b0v; c1[v] = b1v; }
      const _Float16* ap = s_h + aoff * Cmid + asub;
      const v16h* bp = (const v16h*)WbP + (size_t)nt0 * 32 + lane;
      for (int kt = 0; kt < ntk; ++kt) {
        v16h a = frag_a_at(ap);
        c0 = wmma_f16(a, bp[0], c0);
        c1 = wmma_f16(a, bp[boff], c1);
        ap += 32;
        bp += bstep;
      }
#pragma unroll
      for (int v = 0; v < 8; ++v) {
        const int le = v + (grp << 3);
        float* rowp = acc + (size_t)s_dst[le] * Cout + col;
        atomic_add_f32(rowp + nt0 * 16, c0[v]);
        atomic_add_f32(rowp + nt1 * 16, c1[v]);
      }
    }
  }
}

// ---------------------------------------------------------------------------
// Dense node GEMM: Y[T,N] = act(X[T,Cin] @ W^T + b).
// One wave per 16x32 tile pair (A fragment shared across two N tiles).
// grid = (T/16, N/32)
// ---------------------------------------------------------------------------
__global__ void __launch_bounds__(32) k_gemm(
    const _Float16* __restrict__ X, int Cin,
    const _Float16* __restrict__ WP, const float* __restrict__ bias, int N,
    int do_relu, _Float16* __restrict__ out16, float* __restrict__ out32) {
  const int mt = blockIdx.x;
  const int lane = threadIdx.x;
  const int col = lane & 15, grp = lane >> 4;
  const int ntn = N >> 4, ntk = Cin >> 5, half = ntn >> 1;
  const int nt0 = blockIdx.y, nt1 = nt0 + half;

  v8f c0, c1;
  const float b0v = bias[nt0 * 16 + col];
  const float b1v = bias[nt1 * 16 + col];
#pragma unroll
  for (int v = 0; v < 8; ++v) { c0[v] = b0v; c1[v] = b1v; }

  const _Float16* ap = X + (size_t)(mt * 16 + (lane & 15)) * Cin + ((lane >> 4) << 3);
  const v16h* bp = (const v16h*)WP + (size_t)nt0 * 32 + lane;
  const int bstep = ntn * 32, boff = half * 32;
  for (int kt = 0; kt < ntk; ++kt) {
    v16h a = frag_a_at(ap);
    c0 = wmma_f16(a, bp[0], c0);
    c1 = wmma_f16(a, bp[boff], c1);
    ap += 32;
    bp += bstep;
  }
  const int row0 = mt * 16;
#pragma unroll
  for (int v = 0; v < 8; ++v) {
    const int r = row0 + v + (grp << 3);
    float y0 = c0[v], y1 = c1[v];
    if (do_relu) { y0 = fmaxf(y0, 0.0f); y1 = fmaxf(y1, 0.0f); }
    if (out16) {
      out16[(size_t)r * N + nt0 * 16 + col] = (_Float16)y0;
      out16[(size_t)r * N + nt1 * 16 + col] = (_Float16)y1;
    }
    if (out32) {
      out32[(size_t)r * N + nt0 * 16 + col] = y0;
      out32[(size_t)r * N + nt1 * 16 + col] = y1;
    }
  }
}

// ---------------------------------------------------------------------------
// y = LN(relu(acc)) * gamma + beta ; optional (y + addf)*0.5 fused skip.
// One block (C threads) per node; biased variance, eps=1e-5 (torch LayerNorm).
// ---------------------------------------------------------------------------
__global__ void k_relu_ln(const float* __restrict__ acc,
                          const float* __restrict__ gamma,
                          const float* __restrict__ beta, int C,
                          const float* __restrict__ addf,
                          float* __restrict__ out32, _Float16* __restrict__ out16) {
  const int node = blockIdx.x;
  const int t = threadIdx.x;
  __shared__ float s1[256];
  __shared__ float s2[256];
  float v = fmaxf(acc[(size_t)node * C + t], 0.0f);
  s1[t] = v;
  s2[t] = v * v;
  __syncthreads();
  for (int off = C >> 1; off > 0; off >>= 1) {
    if (t < off) { s1[t] += s1[t + off]; s2[t] += s2[t + off]; }
    __syncthreads();
  }
  const float mu  = s1[0] / (float)C;
  const float var = s2[0] / (float)C - mu * mu;
  float y = (v - mu) * rsqrtf(var + 1e-5f) * gamma[t] + beta[t];
  if (addf) y = (y + addf[(size_t)node * C + t]) * 0.5f;
  if (out32) out32[(size_t)node * C + t] = y;
  if (out16) out16[(size_t)node * C + t] = (_Float16)y;
}

// ---------------------------------------------------------------------------
// K[(i*32+a)*256 + (j*32+b)] = sum_h sigma[i,j,h] * blocks[h,a,b]   (O=8,H=16,M=32)
// ---------------------------------------------------------------------------
__global__ void k_koopman(const float* __restrict__ sigma,
                          const float* __restrict__ blocks,
                          float* __restrict__ K) {
  const int idx = blockIdx.x * 256 + threadIdx.x;   // 65536 threads
  const int r = idx >> 8, cc = idx & 255;
  const int i = r >> 5, a = r & 31, j = cc >> 5, b = cc & 31;
  float s = 0.0f;
#pragma unroll
  for (int h = 0; h < 16; ++h)
    s = fmaf(sigma[(i * 8 + j) * 16 + h], blocks[(h * 32 + a) * 32 + b], s);
  K[idx] = s;
}

// Lu[t, j] = sum_u ea[t,u] * Lw[j,u]   (U=4)
__global__ void k_lu(const float* __restrict__ ea, const float* __restrict__ Lw,
                     float* __restrict__ Lu, int rows) {
  const int idx = blockIdx.x * 256 + threadIdx.x;
  if (idx >= rows * 256) return;
  const int t = idx >> 8, j = idx & 255;
  const float* e = ea + t * 4;
  const float* w = Lw + j * 4;
  Lu[idx] = e[0] * w[0] + e[1] * w[1] + e[2] * w[2] + e[3] * w[3];
}

// ---------------------------------------------------------------------------
// Sequential rollout: g_t = g_{t-1} @ K + Lu[t-1]; single 1024-thread block,
// g double-buffered in LDS, K streamed from L2 (coalesced over j), 4-way
// split-K partial sums reduced through LDS. Emits f16 g_hat for decoder gather.
// ---------------------------------------------------------------------------
__global__ void __launch_bounds__(1024) k_rollout(
    const float* __restrict__ g0, const float* __restrict__ K,
    const float* __restrict__ Lu, _Float16* __restrict__ ghat16, int T) {
  __shared__ float gbuf[2][KDIM];
  __shared__ float part[1024];
  const int tid = threadIdx.x;
  const int j = tid & 255, p = tid >> 8;   // 4 partial-sum groups
  if (tid < KDIM) {
    float v = g0[tid];
    gbuf[0][tid] = v;
    ghat16[tid] = (_Float16)v;
  }
  __syncthreads();
  for (int t = 1; t < T; ++t) {
    const float* gin = gbuf[(t - 1) & 1];
    const float* Kp = K + (p * 64) * KDIM + j;
    float s = 0.0f;
#pragma unroll 8
    for (int i = 0; i < 64; ++i) s = fmaf(gin[p * 64 + i], Kp[i * KDIM], s);
    part[tid] = s;
    __syncthreads();
    if (tid < KDIM) {
      float r = part[tid] + part[tid + 256] + part[tid + 512] + part[tid + 768]
              + Lu[(size_t)(t - 1) * KDIM + tid];
      gbuf[t & 1][tid] = r;
      ghat16[(size_t)t * KDIM + tid] = (_Float16)r;
    }
    __syncthreads();
  }
}

// ---------------------------------------------------------------------------
// Host orchestration
// ---------------------------------------------------------------------------
namespace {

struct LinP {
  const float* W;
  const float* b;
  int N, Kin, Kpad;
  _Float16* P;
};

struct GnnP {
  LinP c1a, c1b, c2a, c2b, c3a, c3b, f1, f2, f3, f4;
  const float *n1g, *n1b, *n2g, *n2b, *n3g, *n3b;
  int din, dout;
};

inline int pad32(int x) { return (x + 31) & ~31; }

} // namespace

extern "C" void kernel_launch(void* const* d_in, const int* in_sizes, int n_in,
                              void* d_out_, int out_size, void* d_ws, size_t ws_size,
                              hipStream_t stream) {
  (void)in_sizes; (void)n_in; (void)out_size; (void)ws_size;
  const float* x  = (const float*)d_in[0];
  const int*   ei = (const int*)d_in[1];
  const float* ea = (const float*)d_in[2];
  const int* src = ei;        // edge_index[0]
  const int* dst = ei + NE;   // edge_index[1]
  const float* blocks = (const float*)d_in[55];  // [16,32,32]
  const float* Lw     = (const float*)d_in[56];  // [256,4]
  const float* sigma  = (const float*)d_in[57];  // [8,8,16]
  float* out = (float*)d_out_;

  // pytree leaf order per GNN dict (alphabetical keys):
  // c1a,c1b,c2a,c2b,c3a,c3b,f1,f2,f3,f4 (W,b each) then n1,n2,n3 (g,b each)
  auto make_gnn = [&](int bi, int din, int dout) -> GnnP {
    auto F = [&](int i) -> const float* { return (const float*)d_in[bi + i]; };
    const int dh = 256, dh2 = 128;
    GnnP g{};
    g.din = din; g.dout = dout;
    g.c1a = {F(0),  F(1),  dh,   2 * din + 4, 0, nullptr};
    g.c1b = {F(2),  F(3),  dh,   dh,          0, nullptr};
    g.c2a = {F(4),  F(5),  dh2,  2 * dh + 4,  0, nullptr};
    g.c2b = {F(6),  F(7),  dh2,  dh2,         0, nullptr};
    g.c3a = {F(8),  F(9),  dout, 2 * dh2 + 4, 0, nullptr};
    g.c3b = {F(10), F(11), dout, dout,        0, nullptr};
    g.f1  = {F(12), F(13), dh,   din,         0, nullptr};
    g.f2  = {F(14), F(15), dh,   dh,          0, nullptr};
    g.f3  = {F(16), F(17), dh2,  dh,          0, nullptr};
    g.f4  = {F(18), F(19), dout, dh2,        0, nullptr};
    g.n1g = F(20); g.n1b = F(21);
    g.n2g = F(22); g.n2b = F(23);
    g.n3g = F(24); g.n3b = F(25);
    return g;
  };
  GnnP enc = make_gnn(3, 64, 256);
  GnnP dec = make_gnn(29, 256, 64);

  // bump allocator over workspace
  char* ws = (char*)d_ws;
  size_t off = 0;
  auto alloc = [&](size_t bytes) -> void* {
    off = (off + 255) & ~size_t(255);
    void* p = ws + off;
    off += bytes;
    return p;
  };

  _Float16* x16    = (_Float16*)alloc((size_t)NT * 64 * 2);
  _Float16* ks16   = (_Float16*)alloc((size_t)NT * 256 * 2);
  _Float16* ghat16 = (_Float16*)alloc((size_t)NT * 256 * 2);
  _Float16* h1     = (_Float16*)alloc((size_t)NT * 256 * 2);
  _Float16* h2     = (_Float16*)alloc((size_t)NT * 128 * 2);
  _Float16* t1     = (_Float16*)alloc((size_t)NT * 256 * 2);
  _Float16* t2     = (_Float16*)alloc((size_t)NT * 256 * 2);
  _Float16* t3     = (_Float16*)alloc((size_t)NT * 128 * 2);
  float*    f4o    = (float*)alloc((size_t)NT * 256 * 4);
  float*    acc    = (float*)alloc((size_t)NT * 256 * 4);
  float*    Kmat   = (float*)alloc((size_t)KDIM * KDIM * 4);
  float*    Lu     = (float*)alloc((size_t)(NT - 1) * KDIM * 4);

  auto pack = [&](LinP& L) {
    L.Kpad = pad32(L.Kin);
    L.P = (_Float16*)alloc((size_t)L.Kpad * L.N * 2);
    const int total = L.Kpad * L.N;
    k_pack_w<<<(total + 255) / 256, 256, 0, stream>>>(L.W, L.P, L.N, L.Kin, L.Kpad);
  };
  for (LinP* L : {&enc.c1a, &enc.c1b, &enc.c2a, &enc.c2b, &enc.c3a, &enc.c3b,
                  &enc.f1, &enc.f2, &enc.f3, &enc.f4,
                  &dec.c1a, &dec.c1b, &dec.c2a, &dec.c2b, &dec.c3a, &dec.c3b,
                  &dec.f1, &dec.f2, &dec.f3, &dec.f4})
    pack(*L);

  auto run_gnn = [&](const GnnP& G, const _Float16* xin16,
                     float* out32, _Float16* out16) {
    const int dh = 256, dh2 = 128;
    // conv1 -> relu -> LN
    (void)hipMemsetAsync(acc, 0, (size_t)NT * dh * 4, stream);
    k_edge_conv<<<NE / 16, 128, 0, stream>>>(xin16, G.din, src, dst, ea,
        G.c1a.P, G.c1a.b, G.c1a.Kpad, dh, G.c1b.P, G.c1b.b, dh, acc);
    k_relu_ln<<<NT, dh, 0, stream>>>(acc, G.n1g, G.n1b, dh, nullptr, nullptr, h1);
    // conv2
    (void)hipMemsetAsync(acc, 0, (size_t)NT * dh2 * 4, stream);
    k_edge_conv<<<NE / 16, 128, 0, stream>>>(h1, dh, src, dst, ea,
        G.c2a.P, G.c2a.b, G.c2a.Kpad, dh2, G.c2b.P, G.c2b.b, dh2, acc);
    k_relu_ln<<<NT, dh2, 0, stream>>>(acc, G.n2g, G.n2b, dh2, nullptr, nullptr, h2);
    // conv3 (combine deferred to final relu_ln)
    (void)hipMemsetAsync(acc, 0, (size_t)NT * G.dout * 4, stream);
    k_edge_conv<<<NE / 16, 128, 0, stream>>>(h2, dh2, src, dst, ea,
        G.c3a.P, G.c3a.b, G.c3a.Kpad, G.dout, G.c3b.P, G.c3b.b, G.dout, acc);
    // skip MLP path
    k_gemm<<<dim3(NT / 16, dh / 32), 32, 0, stream>>>(xin16, G.din, G.f1.P, G.f1.b, dh, 1, t1, nullptr);
    k_gemm<<<dim3(NT / 16, dh / 32), 32, 0, stream>>>(t1, dh, G.f2.P, G.f2.b, dh, 1, t2, nullptr);
    k_gemm<<<dim3(NT / 16, dh2 / 32), 32, 0, stream>>>(t2, dh, G.f3.P, G.f3.b, dh2, 1, t3, nullptr);
    k_gemm<<<dim3(NT / 16, G.dout / 32), 32, 0, stream>>>(t3, dh2, G.f4.P, G.f4.b, G.dout, 0, nullptr, f4o);
    // out = (LN(relu(conv3)) + f) * 0.5
    k_relu_ln<<<NT, G.dout, 0, stream>>>(acc, G.n3g, G.n3b, G.dout, f4o, out32, out16);
  };

  // d_out layout: decoded_ae [T,64] | decoded_rollout [T,64] | koopman_states [T,256]
  float* ae_out = out;
  float* ro_out = out + (size_t)NT * 64;
  float* ks_out = out + (size_t)2 * NT * 64;

  k_cvt16<<<(NT * 64 + 255) / 256, 256, 0, stream>>>(x, x16, NT * 64);

  run_gnn(enc, x16, ks_out, ks16);                 // koopman_states (f32 out + f16 copy)

  k_koopman<<<256, 256, 0, stream>>>(sigma, blocks, Kmat);
  k_lu<<<((NT - 1) * 256 + 255) / 256, 256, 0, stream>>>(ea, Lw, Lu, NT - 1);
  k_rollout<<<1, 1024, 0, stream>>>(ks_out, Kmat, Lu, ghat16, NT);

  run_gnn(dec, ks16, ae_out, nullptr);             // decoded_ae
  run_gnn(dec, ghat16, ro_out, nullptr);           // decoded_rollout
}
